// VideoAttention_73529840107718
// MI455X (gfx1250) — compile-verified, matrix operand order checked by reading, not run on hardware
//
#include <hip/hip_runtime.h>

typedef __attribute__((ext_vector_type(16))) __bf16 v16bf;
typedef __attribute__((ext_vector_type(2)))  __bf16 v2bf;
typedef __attribute__((ext_vector_type(8)))  float  v8f;
typedef __attribute__((ext_vector_type(4)))  unsigned int v4u;

#define NSPLIT 8
#define STEPS  (64 / NSPLIT)

// ---------------------------------------------------------------------------
// WMMA helpers (CDNA5 v_wmma_f32_16x16x32_bf16)
// ---------------------------------------------------------------------------
__device__ __forceinline__ v8f wmma_bf16(v16bf a, v16bf b, v8f c) {
  return __builtin_amdgcn_wmma_f32_16x16x32_bf16(false, a, false, b, (short)0, c,
                                                 false, false);
}

// Build a 16-bf16-per-lane WMMA operand from K-major (unit-stride-K) storage.
// Operand VGPR r holds K=(2r+s)+8*half (r<4) / 16+2(r-4)+s+8*half (r>=4),
// i.e. source dwords [4*half, 4*half+4) and [8+4*half, 12+4*half):
// exactly two aligned 128-bit loads. `off` (elements) must be % 8 == 0.
__device__ __forceinline__ v16bf load_op(const __bf16* p, int off, int half) {
  const char* b = (const char*)(p + off) + half * 16;
  union { v4u u[2]; v16bf v; } r;
  r.u[0] = *(const v4u*)(b);        // ds_load_b128 / global_load_b128
  r.u[1] = *(const v4u*)(b + 32);
  return r.v;
}

__device__ __forceinline__ v8f zero8() {
  v8f z = {0.f, 0.f, 0.f, 0.f, 0.f, 0.f, 0.f, 0.f};
  return z;
}

// inv_freq(j) = 10000^(-j/32), j = 0..31
__device__ __forceinline__ float inv_freq(int j) {
  return __expf(-(float)j * (9.2103404f / 32.0f)); // ln(10000)
}

// ---------------------------------------------------------------------------
// CDNA5 async global->LDS DMA (GLOBAL_LOAD_ASYNC_TO_LDS_B128, ASYNCcnt)
// ---------------------------------------------------------------------------
__device__ __forceinline__ unsigned lds_off(const void* p) {
  return (unsigned)(size_t)(const __attribute__((address_space(3))) void*)p;
}
__device__ __forceinline__ void async_cp16(const void* g, unsigned lds_byte) {
  asm volatile("global_load_async_to_lds_b128 %0, %1, off"
               :: "v"(lds_byte), "v"(g) : "memory");
}
__device__ __forceinline__ void wait_async0() {
  asm volatile("s_wait_asynccnt 0" ::: "memory");
}

// ---------------------------------------------------------------------------
// K1: magnitude-preserving weight normalization -> bf16
//     scale = 1 / (||row|| + 1e-4 * sqrt(512))
// ---------------------------------------------------------------------------
__global__ void va_norm_kernel(const float* __restrict__ wqkv,
                               const float* __restrict__ wproj,
                               __bf16* __restrict__ wqkv_bf,
                               __bf16* __restrict__ wproj_bf) {
  __shared__ float red[256];
  int r = blockIdx.x, tid = threadIdx.x;
  const float* src;
  __bf16* dst;
  if (r < 1536) { src = wqkv + (size_t)r * 512;           dst = wqkv_bf + (size_t)r * 512; }
  else          { src = wproj + (size_t)(r - 1536) * 512; dst = wproj_bf + (size_t)(r - 1536) * 512; }
  float2 v = ((const float2*)src)[tid];
  red[tid] = v.x * v.x + v.y * v.y;
  __syncthreads();
  for (int s = 128; s > 0; s >>= 1) {
    if (tid < s) red[tid] += red[tid + s];
    __syncthreads();
  }
  float scale = 1.0f / (sqrtf(red[0]) + 1e-4f * 22.627417f); // sqrt(512)
  v2bf o;
  o[0] = (__bf16)(v.x * scale);
  o[1] = (__bf16)(v.y * scale);
  *(v2bf*)(dst + 2 * tid) = o;
}

// ---------------------------------------------------------------------------
// K2: QKV projection GEMM  Y(1536x256) = Wn(1536x512) x X(512x256)  per batch
//     grid (b, 12 M-blocks of 128, 4 N-blocks of 64), 256 threads = 8 waves.
//     W tile staged by async global->LDS DMA; X staged transposed (K-major).
//     Epilogue: Q -> RoPE(pos 63) * 1/8 -> bf16 ; K,V -> f32 scratch.
// ---------------------------------------------------------------------------
__global__ void va_qkv_kernel(const float* __restrict__ x,
                              const __bf16* __restrict__ wbf,
                              __bf16* __restrict__ qbf,
                              float* __restrict__ knew,
                              float* __restrict__ vnew) {
  __shared__ __bf16 Wl[128 * 32];  // [row][ch-chunk]
  __shared__ __bf16 Xl[64 * 32];   // [pos][ch-chunk]  (transposed)
  int b = blockIdx.x, Mb = blockIdx.y, Nb = blockIdx.z;
  int tid = threadIdx.x, lane = tid & 31, wv = tid >> 5;
  int half = lane >> 4, ln = lane & 15;
  unsigned wl_base = lds_off(Wl);

  v8f acc[4] = {zero8(), zero8(), zero8(), zero8()};

  for (int kc = 0; kc < 16; ++kc) {
    __syncthreads();
    // W tile: 128 rows x 32 ch = 512 x 16B, async DMA straight into LDS
    for (int idx = tid; idx < 512; idx += 256) {
      int rr = idx >> 2, q4 = idx & 3;
      async_cp16(wbf + (size_t)(Mb * 128 + rr) * 512 + kc * 32 + q4 * 8,
                 wl_base + idx * 16);
    }
    // X tile: 32 ch x 64 pos, read f32x4 coalesced, store transposed bf16
    for (int idx = tid; idx < 512; idx += 256) {
      int rr = idx >> 4, c4 = idx & 15; // rr = channel, c4 = pos/4
      float4 xv = ((const float4*)(x + (size_t)(b * 512 + kc * 32 + rr) * 256 +
                                   Nb * 64))[c4];
      Xl[(c4 * 4 + 0) * 32 + rr] = (__bf16)xv.x;
      Xl[(c4 * 4 + 1) * 32 + rr] = (__bf16)xv.y;
      Xl[(c4 * 4 + 2) * 32 + rr] = (__bf16)xv.z;
      Xl[(c4 * 4 + 3) * 32 + rr] = (__bf16)xv.w;
    }
    wait_async0();
    __syncthreads();
    v16bf A = load_op(Wl, (wv * 16 + ln) * 32, half);
    v16bf Bt[4];
#pragma unroll
    for (int nt = 0; nt < 4; ++nt) Bt[nt] = load_op(Xl, (nt * 16 + ln) * 32, half);
#pragma unroll
    for (int nt = 0; nt < 4; ++nt) acc[nt] = wmma_bf16(A, Bt[nt], acc[nt]);
  }

  int rowbase = Mb * 128 + wv * 16;
#pragma unroll
  for (int nt = 0; nt < 4; ++nt) {
    int pos = Nb * 64 + nt * 16 + ln;
#pragma unroll
    for (int v = 0; v < 8; ++v) {
      int o = rowbase + v + 8 * half;
      int qkv = o >> 9, rem = o & 511, head = rem >> 6, ch = rem & 63;
      float val = acc[nt][v];
      if (qkv == 1)      knew[((size_t)(b * 8 + head) * 256 + pos) * 64 + ch] = val;
      else if (qkv == 2) vnew[((size_t)(b * 8 + head) * 256 + pos) * 64 + ch] = val;
    }
    // Q rows: RoPE pairs on adjacent channels (v even / v+1), packed b32 store
#pragma unroll
    for (int v = 0; v < 8; v += 2) {
      int o = rowbase + v + 8 * half;
      if ((o >> 9) == 0) {
        int rem = o & 511, head = rem >> 6, ch = rem & 63; // ch even
        float ang = 63.0f * inv_freq(ch >> 1);
        float sn, cs;
        __sincosf(ang, &sn, &cs);
        float x1 = acc[nt][v], x2 = acc[nt][v + 1];
        v2bf q;
        q[0] = (__bf16)((x1 * cs - x2 * sn) * 0.125f);
        q[1] = (__bf16)((x1 * sn + x2 * cs) * 0.125f);
        *(v2bf*)(qbf + ((size_t)(b * 8 + head) * 256 + pos) * 64 + ch) = q;
      }
    }
  }
}

// ---------------------------------------------------------------------------
// K3: flash attention, split over timesteps.
//     grid (32 bm, NSPLIT), 512 threads = 16 waves (one 16-row Q tile each).
//     LDS (96KB dynamic): Kb[key][ch] bf16 (roped, K-major for QK^T B),
//                         Vt[ch][key] bf16 (transposed, K-major for PV B),
//                         Pb: per-wave [row][key] slab (K-major for PV A).
// ---------------------------------------------------------------------------
__global__ void va_attn_kernel(const float* __restrict__ kcache,
                               const float* __restrict__ vcache,
                               const float* __restrict__ knew,
                               const float* __restrict__ vnew,
                               const __bf16* __restrict__ qbf,
                               float* __restrict__ opart,
                               float* __restrict__ mpart,
                               float* __restrict__ lpart) {
  extern __shared__ char smem[];
  __bf16* Kb = (__bf16*)smem;     // 256 keys * 64 ch
  __bf16* Vt = Kb + 256 * 64;     // 64 ch * 256 keys
  __bf16* Pb = Vt + 64 * 256;     // 16 waves * 16 rows * 64 keys
  int bm = blockIdx.x, split = blockIdx.y;
  int tid = threadIdx.x, lane = tid & 31, wv = tid >> 5;
  int half = lane >> 4, ln = lane & 15;
  __bf16* Pw = Pb + wv * 16 * 64;
  int wrow = wv * 16;

  // Q operands (1/sqrt(64) folded in): two global_load_b128 each
  const __bf16* qp = qbf + ((size_t)bm * 256 + wrow) * 64;
  v16bf qa0 = load_op(qp, ln * 64, half);
  v16bf qa1 = load_op(qp, ln * 64 + 32, half);

  v8f oacc[4] = {zero8(), zero8(), zero8(), zero8()};
  float mrun[8], lrun[8];
#pragma unroll
  for (int v = 0; v < 8; ++v) { mrun[v] = -1e30f; lrun[v] = 0.f; }

  for (int ti = 0; ti < STEPS; ++ti) {
    int t = split * STEPS + ti;
    __syncthreads(); // previous tile fully consumed
    const float* ks = (t < 63) ? kcache + (size_t)(bm * 63 + t) * 256 * 64
                               : knew + (size_t)bm * 256 * 64;
    const float* vs = (t < 63) ? vcache + (size_t)(bm * 63 + t) * 256 * 64
                               : vnew + (size_t)bm * 256 * 64;
    if (ti + 1 < STEPS) { // prefetch next timestep's K tile
      int t2 = t + 1;
      const float* kn = (t2 < 63) ? kcache + (size_t)(bm * 63 + t2) * 256 * 64
                                  : knew + (size_t)bm * 256 * 64;
      __builtin_prefetch(kn + tid * 32, 0, 1);
    }
    // stage K with RoPE(pos t): f32x2 global loads, packed b32 LDS stores
    for (int i = tid; i < 8192; i += 512) {
      int pos = i >> 5, j = i & 31;
      float2 kv = ((const float2*)ks)[pos * 32 + j];
      float ang = (float)t * inv_freq(j);
      float sn, cs;
      __sincosf(ang, &sn, &cs);
      v2bf kk;
      kk[0] = (__bf16)(kv.x * cs - kv.y * sn);
      kk[1] = (__bf16)(kv.x * sn + kv.y * cs);
      *(v2bf*)(Kb + pos * 64 + 2 * j) = kk;
    }
    // stage V transposed: f32x4 global loads, scattered b16 LDS stores
    for (int i = tid; i < 4096; i += 512) {
      int pos = i >> 4, c4 = i & 15;
      float4 vv = ((const float4*)(vs + pos * 64))[c4];
      Vt[(c4 * 4 + 0) * 256 + pos] = (__bf16)vv.x;
      Vt[(c4 * 4 + 1) * 256 + pos] = (__bf16)vv.y;
      Vt[(c4 * 4 + 2) * 256 + pos] = (__bf16)vv.z;
      Vt[(c4 * 4 + 3) * 256 + pos] = (__bf16)vv.w;
    }
    __syncthreads();

    for (int ck = 0; ck < 4; ++ck) { // 64-key chunks
      // batch all 8 B operands first so ds loads pipeline past the wmmas
      v16bf kb[8];
#pragma unroll
      for (int nt = 0; nt < 4; ++nt) {
        int key = ck * 64 + nt * 16 + ln;
        kb[2 * nt]     = load_op(Kb, key * 64, half);      // ch 0..31
        kb[2 * nt + 1] = load_op(Kb, key * 64 + 32, half); // ch 32..63
      }
      v8f s[4];
#pragma unroll
      for (int nt = 0; nt < 4; ++nt) {
        v8f z = zero8();
        z = wmma_bf16(qa0, kb[2 * nt], z);
        z = wmma_bf16(qa1, kb[2 * nt + 1], z);
        s[nt] = z;
      }
      // online softmax (rows in VGPR index v, keys across lanes)
      float psum[8];
#pragma unroll
      for (int v = 0; v < 8; ++v) {
        float c = fmaxf(fmaxf(s[0][v], s[1][v]), fmaxf(s[2][v], s[3][v]));
#pragma unroll
        for (int off = 1; off < 16; off <<= 1) c = fmaxf(c, __shfl_xor(c, off, 32));
        float mn = fmaxf(mrun[v], c);
        float f = __expf(mrun[v] - mn);
        mrun[v] = mn;
        lrun[v] *= f;
#pragma unroll
        for (int nt = 0; nt < 4; ++nt) oacc[nt][v] *= f;
        float ps = 0.f;
#pragma unroll
        for (int nt = 0; nt < 4; ++nt) {
          float p = __expf(s[nt][v] - mn);
          s[nt][v] = p;
          ps += p;
        }
#pragma unroll
        for (int off = 1; off < 16; off <<= 1) ps += __shfl_xor(ps, off, 32);
        psum[v] = ps;
      }
#pragma unroll
      for (int v = 0; v < 8; ++v) lrun[v] += psum[v];
      // C-layout -> A-layout (K-major) via per-wave LDS slab
#pragma unroll
      for (int nt = 0; nt < 4; ++nt)
#pragma unroll
        for (int v = 0; v < 8; ++v)
          Pw[(v + 8 * half) * 64 + nt * 16 + ln] = (__bf16)s[nt][v];
      v16bf pa0 = load_op(Pw, ln * 64, half);
      v16bf pa1 = load_op(Pw, ln * 64 + 32, half);
      v16bf vb[8];
#pragma unroll
      for (int nt = 0; nt < 4; ++nt) {
        vb[2 * nt]     = load_op(Vt, (nt * 16 + ln) * 256 + ck * 64, half);
        vb[2 * nt + 1] = load_op(Vt, (nt * 16 + ln) * 256 + ck * 64 + 32, half);
      }
#pragma unroll
      for (int nt = 0; nt < 4; ++nt) {
        oacc[nt] = wmma_bf16(pa0, vb[2 * nt], oacc[nt]);
        oacc[nt] = wmma_bf16(pa1, vb[2 * nt + 1], oacc[nt]);
      }
    }
  }
  // write split-partial O (un-normalized) + stats
  size_t base = (size_t)(split * 32 + bm) * 256;
#pragma unroll
  for (int nt = 0; nt < 4; ++nt)
#pragma unroll
    for (int v = 0; v < 8; ++v) {
      int row = wrow + v + 8 * half;
      opart[(base + row) * 64 + nt * 16 + ln] = oacc[nt][v];
    }
  if (ln == 0) {
#pragma unroll
    for (int v = 0; v < 8; ++v) {
      int row = wrow + v + 8 * half;
      mpart[base + row] = mrun[v];
      lpart[base + row] = lrun[v];
    }
  }
}

// ---------------------------------------------------------------------------
// K4a: merge split partials, normalize, write interleaved-channel activation
//      Ocomb[b][ch*8+head][pos] (bf16) for the output projection.
// ---------------------------------------------------------------------------
__global__ void va_combine_kernel(const float* __restrict__ opart,
                                  const float* __restrict__ mpart,
                                  const float* __restrict__ lpart,
                                  __bf16* __restrict__ ocomb) {
  int bm = blockIdx.x, row = threadIdx.x;
  int b = bm >> 3, head = bm & 7;
  float mv[NSPLIT], M = -1e30f;
  for (int s = 0; s < NSPLIT; ++s) {
    mv[s] = mpart[(size_t)(s * 32 + bm) * 256 + row];
    M = fmaxf(M, mv[s]);
  }
  float w[NSPLIT], l = 0.f;
  for (int s = 0; s < NSPLIT; ++s) {
    w[s] = __expf(mv[s] - M);
    l += w[s] * lpart[(size_t)(s * 32 + bm) * 256 + row];
  }
  float inv = 1.0f / l;
  for (int ch = 0; ch < 64; ++ch) {
    float a = 0.f;
    for (int s = 0; s < NSPLIT; ++s)
      a += w[s] * opart[((size_t)(s * 32 + bm) * 256 + row) * 64 + ch];
    ocomb[((size_t)b * 512 + ch * 8 + head) * 256 + row] = (__bf16)(a * inv);
  }
}

// ---------------------------------------------------------------------------
// K4b: output projection GEMM + mp_sum epilogue.
//      out = x * (0.7/sqrt(0.58)) + proj * (0.3/sqrt(0.58))
// ---------------------------------------------------------------------------
__global__ void va_proj_kernel(const __bf16* __restrict__ ocomb,
                               const __bf16* __restrict__ wbf,
                               const float* __restrict__ x,
                               float* __restrict__ out) {
  __shared__ __bf16 Wl[128 * 32]; // [row][ch-chunk]
  __shared__ __bf16 Xl[64 * 32];  // [pos][ch-chunk] (transposed)
  int b = blockIdx.x, Mb = blockIdx.y, Nb = blockIdx.z;
  int tid = threadIdx.x, lane = tid & 31, wv = tid >> 5;
  int half = lane >> 4, ln = lane & 15;
  unsigned wl_base = lds_off(Wl);

  v8f acc[4] = {zero8(), zero8(), zero8(), zero8()};

  for (int kc = 0; kc < 16; ++kc) {
    __syncthreads();
    // W tile via async global->LDS DMA
    for (int idx = tid; idx < 512; idx += 256) {
      int rr = idx >> 2, q4 = idx & 3;
      async_cp16(wbf + (size_t)(Mb * 128 + rr) * 512 + kc * 32 + q4 * 8,
                 wl_base + idx * 16);
    }
    // ocomb tile: 32 ch x 64 pos, b128 reads, store transposed
    {
      int rr = tid >> 3, q4 = tid & 7; // rr = channel, q4 selects 8 pos
      union { v4u u; __bf16 h[8]; } d;
      d.u = ((const v4u*)(ocomb + (size_t)(b * 512 + kc * 32 + rr) * 256 +
                          Nb * 64))[q4];
#pragma unroll
      for (int k = 0; k < 8; ++k) Xl[(q4 * 8 + k) * 32 + rr] = d.h[k];
    }
    wait_async0();
    __syncthreads();
    v16bf A = load_op(Wl, (wv * 16 + ln) * 32, half);
    v16bf Bt[4];
#pragma unroll
    for (int nt = 0; nt < 4; ++nt) Bt[nt] = load_op(Xl, (nt * 16 + ln) * 32, half);
#pragma unroll
    for (int nt = 0; nt < 4; ++nt) acc[nt] = wmma_bf16(A, Bt[nt], acc[nt]);
  }

  int rowbase = Mb * 128 + wv * 16;
#pragma unroll
  for (int nt = 0; nt < 4; ++nt) {
    int pos = Nb * 64 + nt * 16 + ln;
#pragma unroll
    for (int v = 0; v < 8; ++v) {
      int o = rowbase + v + 8 * half;
      size_t idx = (size_t)(b * 512 + o) * 256 + pos;
      out[idx] = x[idx] * 0.9191450f + acc[nt][v] * 0.3939193f;
    }
  }
}

// ---------------------------------------------------------------------------
extern "C" void kernel_launch(void* const* d_in, const int* in_sizes, int n_in,
                              void* d_out, int out_size, void* d_ws, size_t ws_size,
                              hipStream_t stream) {
  (void)in_sizes; (void)n_in; (void)out_size; (void)ws_size;
  const float* x     = (const float*)d_in[0];
  const float* ck    = (const float*)d_in[1];
  const float* cv    = (const float*)d_in[2];
  const float* wqkv  = (const float*)d_in[3];
  const float* wproj = (const float*)d_in[4];
  float* out = (float*)d_out;
  char* ws = (char*)d_ws;

  __bf16* wqkv_bf  = (__bf16*)(ws + 0);                    // 1,572,864 B
  __bf16* wproj_bf = (__bf16*)(ws + 1572864);              //   524,288 B
  __bf16* qbf      = (__bf16*)(ws + 2097152);              // 1,048,576 B
  float*  knew     = (float*)(ws + 3145728);               // 2,097,152 B
  float*  vnew     = (float*)(ws + 5242880);               // 2,097,152 B
  float*  opart    = (float*)(ws + 7340032);               // 16,777,216 B
  float*  mpart    = (float*)(ws + 24117248);              //   262,144 B
  float*  lpart    = (float*)(ws + 24379392);              //   262,144 B
  __bf16* ocomb    = (__bf16*)(ws + 24641536);             // 1,048,576 B (total ~25.7 MB)

  va_norm_kernel<<<2048, 256, 0, stream>>>(wqkv, wproj, wqkv_bf, wproj_bf);
  va_qkv_kernel<<<dim3(4, 12, 4), 256, 0, stream>>>(x, wqkv_bf, qbf, knew, vnew);
  va_attn_kernel<<<dim3(32, NSPLIT), 512, 98304, stream>>>(ck, cv, knew, vnew, qbf,
                                                           opart, mpart, lpart);
  va_combine_kernel<<<32, 256, 0, stream>>>(opart, mpart, lpart, ocomb);
  va_proj_kernel<<<dim3(4, 4, 4), 256, 0, stream>>>(ocomb, wproj_bf, x, out);
}
